// RelPosDecaySelfAttn_69956427317532
// MI455X (gfx1250) — compile-verified, matrix operand order checked by reading, not run on hardware
//
#include <hip/hip_runtime.h>

#define D_MODEL 512
#define NHEAD   8
#define D_HEAD  64
#define BATCH   4
#define SEQ     2048
#define CUT_IDX 1024.0f
#define ALPHA   0.02f
#define BETA    0.5f

typedef __attribute__((ext_vector_type(16))) _Float16 v16h;
typedef __attribute__((ext_vector_type(8)))  _Float16 v8h;
typedef __attribute__((ext_vector_type(8)))  float    v8f;

// ---------------------------------------------------------------------------
// Fragment loaders for V_WMMA_F32_16X16X32_F16 (wave32).
// 16-bit A (16x32 MxK) layout:  lane L(0..15) row M=L, holds K={0..7,16..23};
// lane L+16 row M=L, holds K={8..15,24..31}.  B (32x16 KxN) is the same
// pattern applied to B^T (NxK), i.e. lane = N%16.
// ---------------------------------------------------------------------------
__device__ __forceinline__ v16h load_frag_f16(const _Float16* __restrict__ base,
                                              int row0, int ld, int k0) {
  const int lane = threadIdx.x & 31;
  const int r    = lane & 15;
  const int half = lane >> 4;
  const _Float16* p = base + (size_t)(row0 + r) * ld + k0 + half * 8;
  v8h lo = *(const v8h*)(p);
  v8h hi = *(const v8h*)(p + 16);
  return __builtin_shufflevector(lo, hi, 0,1,2,3,4,5,6,7,8,9,10,11,12,13,14,15);
}

__device__ __forceinline__ v16h load_frag_f32_cvt(const float* __restrict__ base,
                                                  int row0, int ld, int k0) {
  const int lane = threadIdx.x & 31;
  const int r    = lane & 15;
  const int half = lane >> 4;
  const float* p = base + (size_t)(row0 + r) * ld + k0 + half * 8;
  const float4* p4 = (const float4*)p;
  float4 a0 = p4[0], a1 = p4[1];   // K offsets +0..7
  float4 a2 = p4[4], a3 = p4[5];   // K offsets +16..23
  v16h out;
  out[0]=(_Float16)a0.x; out[1]=(_Float16)a0.y; out[2]=(_Float16)a0.z; out[3]=(_Float16)a0.w;
  out[4]=(_Float16)a1.x; out[5]=(_Float16)a1.y; out[6]=(_Float16)a1.z; out[7]=(_Float16)a1.w;
  out[8]=(_Float16)a2.x; out[9]=(_Float16)a2.y; out[10]=(_Float16)a2.z; out[11]=(_Float16)a2.w;
  out[12]=(_Float16)a3.x; out[13]=(_Float16)a3.y; out[14]=(_Float16)a3.z; out[15]=(_Float16)a3.w;
  return out;
}

#define WMMA_F16(A,B,C) __builtin_amdgcn_wmma_f32_16x16x32_f16(false,(A),false,(B),(short)0,(C),false,false)

// ---------------------------------------------------------------------------
// K0: f32 -> f16 conversion
// ---------------------------------------------------------------------------
__global__ void cvt_f32_to_f16(const float* __restrict__ src,
                               _Float16* __restrict__ dst, int n) {
  int i = blockIdx.x * 256 + threadIdx.x;
  if (i < n) dst[i] = (_Float16)src[i];
}

// ---------------------------------------------------------------------------
// K1: fused QKV projection.  C[8192x512] = x16 @ W^T + b
// grid (512, 8, 3); block 32 (one wave computes a 16x64 tile)
// z: 0->q (scaled by 1/8, layout [b][h][n][d]); 1->k ([b][h][n][d]);
//    2->v transposed ([b][h][d][n])
// ---------------------------------------------------------------------------
__global__ void qkv_gemm(const _Float16* __restrict__ x16,
                         const _Float16* __restrict__ wq,
                         const _Float16* __restrict__ wk,
                         const _Float16* __restrict__ wv,
                         const float* __restrict__ bq,
                         const float* __restrict__ bk,
                         const float* __restrict__ bv,
                         _Float16* __restrict__ q16,
                         _Float16* __restrict__ k16,
                         _Float16* __restrict__ vT16) {
  const int lane  = threadIdx.x & 31;
  const int row0  = blockIdx.x * 16;       // token-tile
  const int nb    = blockIdx.y * 64;       // output-feature base
  const int which = blockIdx.z;
  const _Float16* W    = (which == 0) ? wq : (which == 1) ? wk : wv;
  const float*    bias = (which == 0) ? bq : (which == 1) ? bk : bv;

  v8f acc[4] = {};
  for (int k0 = 0; k0 < D_MODEL; k0 += 32) {
    v16h a = load_frag_f16(x16, row0, D_MODEL, k0);
#pragma unroll
    for (int t = 0; t < 4; ++t) {
      v16h bf = load_frag_f16(W, nb + t * 16, D_MODEL, k0);
      acc[t] = WMMA_F16(a, bf, acc[t]);
    }
  }

  const int col = lane & 15;
  const int rb  = (lane >> 4) * 8;
#pragma unroll
  for (int t = 0; t < 4; ++t) {
    const int nf = nb + t * 16 + col;
    const float bval = bias[nf];
    const int h = nf >> 6, d = nf & 63;
#pragma unroll
    for (int e = 0; e < 8; ++e) {
      const int m  = row0 + rb + e;        // global token row
      const int bi = m >> 11;              // batch
      const int n  = m & (SEQ - 1);
      const float v = acc[t][e] + bval;
      if (which == 0)
        q16[((size_t)(bi * NHEAD + h) * SEQ + n) * D_HEAD + d] = (_Float16)(v * 0.125f);
      else if (which == 1)
        k16[((size_t)(bi * NHEAD + h) * SEQ + n) * D_HEAD + d] = (_Float16)v;
      else
        vT16[((size_t)(bi * NHEAD + h) * D_HEAD + d) * SEQ + n] = (_Float16)v;
    }
  }
}

// ---------------------------------------------------------------------------
// K2: logits + bias + softmax.  One wave owns 16 query rows of one (b,h).
// Sweep 1: q.k^T (2 WMMA / 16x16 tile), bias, exp, row-sum, store unnorm P.
// Sweep 2: normalize in place (strip is L2-resident).
// grid (128, 8, 4); block 32
// ---------------------------------------------------------------------------
__global__ void attn_softmax(const _Float16* __restrict__ q16,
                             const _Float16* __restrict__ k16,
                             const unsigned char* __restrict__ mask,
                             float* __restrict__ attnOut) {
  const int lane = threadIdx.x & 31;
  const int row0 = blockIdx.x * 16;
  const int h = blockIdx.y, b = blockIdx.z;
  const size_t headOff = (size_t)(b * NHEAD + h) * SEQ * D_HEAD;
  const _Float16* q = q16 + headOff;
  const _Float16* k = k16 + headOff;
  float* A = attnOut + (size_t)(b * NHEAD + h) * SEQ * SEQ;

  const v16h aq0 = load_frag_f16(q, row0, D_HEAD, 0);
  const v16h aq1 = load_frag_f16(q, row0, D_HEAD, 32);

  const int col = lane & 15;
  const int rb  = (lane >> 4) * 8;
  float lsum[8] = {0.f,0.f,0.f,0.f,0.f,0.f,0.f,0.f};

  for (int jt = 0; jt < SEQ / 16; ++jt) {
    v16h b0 = load_frag_f16(k, jt * 16, D_HEAD, 0);
    v16h b1 = load_frag_f16(k, jt * 16, D_HEAD, 32);
    v8f acc = {};
    acc = WMMA_F16(aq0, b0, acc);
    acc = WMMA_F16(aq1, b1, acc);

    const int j = jt * 16 + col;
    const float colBias = BETA * fabsf((float)j - CUT_IDX);
    const bool masked = mask[b * SEQ + j] != 0;
#pragma unroll
    for (int e = 0; e < 8; ++e) {
      const int i = row0 + rb + e;
      float s = acc[e] - ALPHA * fabsf((float)(i - j)) - colBias;
      float p = masked ? 0.0f : __expf(fminf(s, 60.0f));
      lsum[e] += p;
      A[(size_t)i * SEQ + j] = p;
    }
  }

  // cross-lane row sums: rows live in 16-lane halves; xor masks < 16 stay in-half
#pragma unroll
  for (int e = 0; e < 8; ++e) {
    float v = lsum[e];
    v += __shfl_xor(v, 1, 32);
    v += __shfl_xor(v, 2, 32);
    v += __shfl_xor(v, 4, 32);
    v += __shfl_xor(v, 8, 32);
    lsum[e] = 1.0f / v;
  }

  for (int jt = 0; jt < SEQ / 16; ++jt) {
    const int j = jt * 16 + col;
#pragma unroll
    for (int e = 0; e < 8; ++e) {
      const int i = row0 + rb + e;
      const size_t idx = (size_t)i * SEQ + j;
      A[idx] = A[idx] * lsum[e];
    }
  }
}

// ---------------------------------------------------------------------------
// K3: ctx = attn @ v.   A = attn (f32 -> f16 in-register), B = v^T (f16).
// grid (128, 8, 4); block 32; wave computes 16 x 64, K loop over 2048.
// ctx layout: [b][n][h*64+d] (f16) so out-proj reads row-major.
// ---------------------------------------------------------------------------
__global__ void attn_v(const float* __restrict__ attn,
                       const _Float16* __restrict__ vT16,
                       _Float16* __restrict__ ctx16) {
  const int lane = threadIdx.x & 31;
  const int row0 = blockIdx.x * 16;
  const int h = blockIdx.y, b = blockIdx.z;
  const float* A = attn + (size_t)(b * NHEAD + h) * SEQ * SEQ;
  const _Float16* vT = vT16 + (size_t)(b * NHEAD + h) * D_HEAD * SEQ;

  v8f acc[4] = {};
  for (int k0 = 0; k0 < SEQ; k0 += 32) {
    v16h a = load_frag_f32_cvt(A, row0, SEQ, k0);
#pragma unroll
    for (int t = 0; t < 4; ++t) {
      v16h bf = load_frag_f16(vT, t * 16, SEQ, k0);
      acc[t] = WMMA_F16(a, bf, acc[t]);
    }
  }

  const int col = lane & 15;
  const int rb  = (lane >> 4) * 8;
#pragma unroll
  for (int t = 0; t < 4; ++t) {
    const int d = t * 16 + col;
#pragma unroll
    for (int e = 0; e < 8; ++e) {
      const int i = row0 + rb + e;
      ctx16[(size_t)(b * SEQ + i) * D_MODEL + h * D_HEAD + d] = (_Float16)acc[t][e];
    }
  }
}

// ---------------------------------------------------------------------------
// K4: out = ctx @ Wo^T + bo   (f32 output into d_out's `out` region)
// grid (512, 8); block 32
// ---------------------------------------------------------------------------
__global__ void out_proj(const _Float16* __restrict__ ctx16,
                         const _Float16* __restrict__ wo16,
                         const float* __restrict__ bo,
                         float* __restrict__ out) {
  const int lane = threadIdx.x & 31;
  const int row0 = blockIdx.x * 16;
  const int nb   = blockIdx.y * 64;

  v8f acc[4] = {};
  for (int k0 = 0; k0 < D_MODEL; k0 += 32) {
    v16h a = load_frag_f16(ctx16, row0, D_MODEL, k0);
#pragma unroll
    for (int t = 0; t < 4; ++t) {
      v16h bf = load_frag_f16(wo16, nb + t * 16, D_MODEL, k0);
      acc[t] = WMMA_F16(a, bf, acc[t]);
    }
  }

  const int col = lane & 15;
  const int rb  = (lane >> 4) * 8;
#pragma unroll
  for (int t = 0; t < 4; ++t) {
    const int nf = nb + t * 16 + col;
    const float bval = bo[nf];
#pragma unroll
    for (int e = 0; e < 8; ++e) {
      const int m = row0 + rb + e;
      out[(size_t)m * D_MODEL + nf] = acc[t][e] + bval;
    }
  }
}

// ---------------------------------------------------------------------------
extern "C" void kernel_launch(void* const* d_in, const int* in_sizes, int n_in,
                              void* d_out, int out_size, void* d_ws, size_t ws_size,
                              hipStream_t stream) {
  const float* x   = (const float*)d_in[0];
  const unsigned char* mask = (const unsigned char*)d_in[1];
  const float* Wq  = (const float*)d_in[2];
  const float* bq  = (const float*)d_in[3];
  const float* Wk  = (const float*)d_in[4];
  const float* bk  = (const float*)d_in[5];
  const float* Wv  = (const float*)d_in[6];
  const float* bv  = (const float*)d_in[7];
  const float* Wo  = (const float*)d_in[8];
  const float* bo  = (const float*)d_in[9];

  float* outPtr  = (float*)d_out;                                    // [B,N,D]
  float* attnPtr = outPtr + (size_t)BATCH * SEQ * D_MODEL;           // [B,H,N,N]

  // workspace layout (f16 buffers), all offsets 256B-aligned
  char* ws = (char*)d_ws;
  const size_t NX  = (size_t)BATCH * SEQ * D_MODEL;   // 4,194,304
  const size_t NW  = (size_t)D_MODEL * D_MODEL;       // 262,144
  _Float16* x16  = (_Float16*)(ws);                         // 8 MB
  _Float16* wq16 = (_Float16*)(ws + NX * 2);
  _Float16* wk16 = wq16 + NW;
  _Float16* wv16 = wk16 + NW;
  _Float16* wo16 = wv16 + NW;
  _Float16* q16  = wo16 + NW;                               // 8 MB
  _Float16* k16  = q16 + NX;                                // 8 MB
  _Float16* vT16 = k16 + NX;                                // 8 MB
  _Float16* ctx16 = vT16 + NX;                              // 8 MB

  // K0: downconvert activations & weights
  cvt_f32_to_f16<<<(int)((NX + 255) / 256), 256, 0, stream>>>(x,  x16,  (int)NX);
  cvt_f32_to_f16<<<(int)((NW + 255) / 256), 256, 0, stream>>>(Wq, wq16, (int)NW);
  cvt_f32_to_f16<<<(int)((NW + 255) / 256), 256, 0, stream>>>(Wk, wk16, (int)NW);
  cvt_f32_to_f16<<<(int)((NW + 255) / 256), 256, 0, stream>>>(Wv, wv16, (int)NW);
  cvt_f32_to_f16<<<(int)((NW + 255) / 256), 256, 0, stream>>>(Wo, wo16, (int)NW);

  // K1: QKV projections
  qkv_gemm<<<dim3(BATCH * SEQ / 16, D_MODEL / 64, 3), 32, 0, stream>>>(
      x16, wq16, wk16, wv16, bq, bk, bv, q16, k16, vT16);

  // K2: logits + bias + softmax -> attn (f32, in d_out)
  attn_softmax<<<dim3(SEQ / 16, NHEAD, BATCH), 32, 0, stream>>>(
      q16, k16, mask, attnPtr);

  // K3: ctx = attn @ v
  attn_v<<<dim3(SEQ / 16, NHEAD, BATCH), 32, 0, stream>>>(attnPtr, vT16, ctx16);

  // K4: out = ctx @ Wo^T + bo
  out_proj<<<dim3(BATCH * SEQ / 16, D_MODEL / 64), 32, 0, stream>>>(
      ctx16, wo16, bo, outPtr);
}
